// SwinBlock_23132693856852
// MI455X (gfx1250) — compile-verified, matrix operand order checked by reading, not run on hardware
//
#include <hip/hip_runtime.h>
#include <hip/hip_bf16.h>
#include <math.h>

typedef _Float16 v16h __attribute__((ext_vector_type(16)));
typedef _Float16 v8h  __attribute__((ext_vector_type(8)));
typedef float    v8f  __attribute__((ext_vector_type(8)));

#define DIMC   192
#define NHEADS 6
#define HD     32
#define WSZ    7
#define WS2    49
#define HW     112
#define MLPD   768
#define NEGV   (-1e9f)
#define EPSV   1e-5f

// ---- fragment helpers (CDNA5 wave32 WMMA f16 16x16x32 layout) ----
// A 16x32 and B(as N-major, i.e. B^T row-major NxK) share the same mapping:
// lane half = lane>>4, row = base + (lane&15);
// elements 0..7  = K  kbase + 8*half + 0..7
// elements 8..15 = K  kbase + 8*half + 16..23
__device__ __forceinline__ v16h ld_frag16(const _Float16* p) {
  v8h a = *(const v8h*)(p);
  v8h b = *(const v8h*)(p + 16);
  v16h r;
#pragma unroll
  for (int i = 0; i < 8; ++i) { r[i] = a[i]; r[i + 8] = b[i]; }
  return r;
}
__device__ __forceinline__ v16h frag_rm(const _Float16* base, int rbase,
                                        int stride, int kbase, int lane) {
  int half = lane >> 4;
  int row  = rbase + (lane & 15);
  return ld_frag16(base + (size_t)row * stride + kbase + half * 8);
}
__device__ __forceinline__ v8f wmma32(v16h a, v16h b, v8f c) {
  return __builtin_amdgcn_wmma_f32_16x16x32_f16(false, a, false, b,
                                                (short)0, c, false, false);
}
__device__ __forceinline__ v8f vzero8() {
  v8f z;
#pragma unroll
  for (int i = 0; i < 8; ++i) z[i] = 0.f;
  return z;
}

// ---------------- prep: f16-transpose weights (N x K), rel-pos bias table ----
__global__ void swin_prep_kernel(const float* __restrict__ wqkv,
                                 const float* __restrict__ wout,
                                 const float* __restrict__ w1,
                                 const float* __restrict__ w2,
                                 const float* __restrict__ pos,
                                 _Float16* __restrict__ wqkvT,
                                 _Float16* __restrict__ woutT,
                                 _Float16* __restrict__ w1T,
                                 _Float16* __restrict__ w2T,
                                 float* __restrict__ bias49) {
  int i = blockIdx.x * blockDim.x + threadIdx.x;
  if (i < 576 * 192) { int n = i / 192, k = i % 192; wqkvT[i] = (_Float16)wqkv[k * 576 + n]; return; }
  i -= 576 * 192;
  if (i < 192 * 192) { int n = i / 192, k = i % 192; woutT[i] = (_Float16)wout[k * 192 + n]; return; }
  i -= 192 * 192;
  if (i < 768 * 192) { int n = i / 192, k = i % 192; w1T[i] = (_Float16)w1[k * 768 + n]; return; }
  i -= 768 * 192;
  if (i < 192 * 768) { int n = i / 768, k = i % 768; w2T[i] = (_Float16)w2[k * 192 + n]; return; }
  i -= 192 * 768;
  if (i < WS2 * WS2) {
    int a = i / WS2, b = i % WS2;
    int ri = a / WSZ, ci = a % WSZ, rj = b / WSZ, cj = b % WSZ;
    bias49[i] = pos[(ri - rj + WSZ - 1) * (2 * WSZ - 1) + (ci - cj + WSZ - 1)];
  }
}

// ---------------- fused: shift + LN1 + QKV + attention + out-proj + residual --
__global__ __launch_bounds__(256) void swin_attn_kernel(
    const float* __restrict__ x, const _Float16* __restrict__ wqkvT,
    const _Float16* __restrict__ woutT, const float* __restrict__ b_out,
    const float* __restrict__ bias49, const float* __restrict__ ln_g,
    const float* __restrict__ ln_b, float* __restrict__ out) {
  __shared__ __align__(16) _Float16 sA[64 * 200];    // LN'd window tokens, f16
  __shared__ __align__(16) _Float16 sQ[64 * 40];
  __shared__ __align__(16) _Float16 sK[64 * 40];
  __shared__ __align__(16) unsigned char sU[64 * 52 * 4]; // S(f32) / Vt(f16) overlay
  __shared__ __align__(16) _Float16 sP[64 * 72];
  __shared__ __align__(16) _Float16 sO[64 * 40];
  float*    sS  = (float*)sU;
  _Float16* sVt = (_Float16*)sU;

  const int tid  = threadIdx.x;
  const int lane = tid & 31;
  const int wv   = tid >> 5;
  const int half = lane >> 4;
  const int win  = blockIdx.x;         // b*256 + wy*16 + wx
  const int b    = win >> 8;
  const int wy   = (win >> 4) & 15;
  const int wx   = win & 15;

  // ---- shifted gather + LayerNorm1 into sA (rows 49..63 zero) ----
  if (tid < 64) {
    int m = tid;
    if (m < WS2) {
      int r = m / WSZ, c = m % WSZ;
      int h = (wy * WSZ + r + 3) % HW;
      int w = (wx * WSZ + c + 3) % HW;
      const float* px = x + ((size_t)((b * HW + h) * HW + w)) * DIMC;
      float mu = 0.f;
#pragma unroll
      for (int k = 0; k < DIMC; k += 4) {
        float4 v = *(const float4*)(px + k);
        mu += v.x + v.y + v.z + v.w;
      }
      mu *= (1.f / DIMC);
      float var = 0.f;
#pragma unroll
      for (int k = 0; k < DIMC; k += 4) {
        float4 v = *(const float4*)(px + k);
        float a0 = v.x - mu, a1 = v.y - mu, a2 = v.z - mu, a3 = v.w - mu;
        var += a0 * a0 + a1 * a1 + a2 * a2 + a3 * a3;
      }
      var *= (1.f / DIMC);
      float rs = rsqrtf(var + EPSV);
      for (int k = 0; k < DIMC; ++k)
        sA[m * 200 + k] = (_Float16)((px[k] - mu) * rs * ln_g[k] + ln_b[k]);
    } else {
      for (int k = 0; k < DIMC; ++k) sA[m * 200 + k] = (_Float16)0.f;
    }
  }
  __syncthreads();

  v8f cout[6];
#pragma unroll
  for (int j = 0; j < 6; ++j) cout[j] = vzero8();

#pragma unroll 1
  for (int head = 0; head < NHEADS; ++head) {
    // ---- Q and K GEMMs (16 wave-tiles: 4 Mtiles x 2 Ntiles x {Q,K}) ----
#pragma unroll
    for (int t2 = 0; t2 < 2; ++t2) {
      int task = wv + t2 * 8;
      int isK  = task >> 3;
      int tt   = task & 7;
      int mt   = tt >> 1, nt = tt & 1;
      int nbase = isK * DIMC + head * HD + nt * 16;
      v8f acc = vzero8();
#pragma unroll
      for (int ks = 0; ks < 6; ++ks) {
        v16h fa = frag_rm(sA, mt * 16, 200, ks * 32, lane);
        v16h fb = frag_rm(wqkvT, nbase, DIMC, ks * 32, lane);
        acc = wmma32(fa, fb, acc);
      }
      _Float16* dst = isK ? sK : sQ;
      int n = nt * 16 + (lane & 15);
#pragma unroll
      for (int r = 0; r < 8; ++r) {
        int m = mt * 16 + r + half * 8;
        dst[m * 40 + n] = (_Float16)acc[r];
      }
    }
    __syncthreads();

    // ---- S = Q K^T  (single K=32 step), store f32 ----
#pragma unroll
    for (int t2 = 0; t2 < 2; ++t2) {
      int task = wv + t2 * 8;
      int mt = task >> 2, nt = task & 3;
      v8f acc = vzero8();
      v16h fa = frag_rm(sQ, mt * 16, 40, 0, lane);
      v16h fb = frag_rm(sK, nt * 16, 40, 0, lane);
      acc = wmma32(fa, fb, acc);
      int n = nt * 16 + (lane & 15);
#pragma unroll
      for (int r = 0; r < 8; ++r) {
        int m = mt * 16 + r + half * 8;
        sS[m * 52 + n] = acc[r];
      }
    }
    __syncthreads();

    // ---- softmax rows (scale + rel-bias + analytic shift masks) ----
    if (tid < 64) {
      int i = tid;
      if (i < WS2) {
        const float scale = 0.17677669529663687f; // 32^-0.5
        const bool mrow = (wy == 15), mcol = (wx == 15);
        float mx = -1e30f;
        for (int j = 0; j < WS2; ++j) {
          float v = sS[i * 52 + j] * scale + bias49[i * WS2 + j];
          if (mrow && (((i >= 28) ? 1 : 0) ^ ((j >= 28) ? 1 : 0))) v += NEGV;
          if (mcol && ((((i % WSZ) >= 4) ? 1 : 0) ^ (((j % WSZ) >= 4) ? 1 : 0))) v += NEGV;
          sS[i * 52 + j] = v;
          mx = fmaxf(mx, v);
        }
        float sum = 0.f;
        for (int j = 0; j < WS2; ++j) {
          float e = __expf(sS[i * 52 + j] - mx);
          sS[i * 52 + j] = e;
          sum += e;
        }
        float inv = 1.f / sum;
        for (int j = 0; j < WS2; ++j) sP[i * 72 + j] = (_Float16)(sS[i * 52 + j] * inv);
        for (int j = WS2; j < 64; ++j) sP[i * 72 + j] = (_Float16)0.f;
      } else {
        for (int j = 0; j < 64; ++j) sP[i * 72 + j] = (_Float16)0.f;
      }
    }
    __syncthreads();

    // ---- V GEMM -> Vt (transposed, overlays sS) ----
    {
      int mt = wv >> 1, nt = wv & 1;
      int nbase = 2 * DIMC + head * HD + nt * 16;
      v8f acc = vzero8();
#pragma unroll
      for (int ks = 0; ks < 6; ++ks) {
        v16h fa = frag_rm(sA, mt * 16, 200, ks * 32, lane);
        v16h fb = frag_rm(wqkvT, nbase, DIMC, ks * 32, lane);
        acc = wmma32(fa, fb, acc);
      }
      int d = nt * 16 + (lane & 15);
#pragma unroll
      for (int r = 0; r < 8; ++r) {
        int m = mt * 16 + r + half * 8;
        sVt[d * 72 + m] = (_Float16)acc[r];
      }
    }
    __syncthreads();

    // ---- O = P @ V (A=P rows, B=Vt rows=head-dim) ----
    {
      int mt = wv >> 1, nt = wv & 1;
      v8f acc = vzero8();
#pragma unroll
      for (int ks = 0; ks < 2; ++ks) {
        v16h fa = frag_rm(sP, mt * 16, 72, ks * 32, lane);
        v16h fb = frag_rm(sVt, nt * 16, 72, ks * 32, lane);
        acc = wmma32(fa, fb, acc);
      }
      int d = nt * 16 + (lane & 15);
#pragma unroll
      for (int r = 0; r < 8; ++r) {
        int m = mt * 16 + r + half * 8;
        sO[m * 40 + d] = (_Float16)acc[r];
      }
    }
    __syncthreads();

    // ---- out-projection: accumulate this head's K=32 slice ----
    {
      int mt  = wv >> 1;
      int ntb = (wv & 1) * 6;
      v16h fa = frag_rm(sO, mt * 16, 40, 0, lane);
#pragma unroll
      for (int j = 0; j < 6; ++j) {
        v16h fb = frag_rm(woutT, (ntb + j) * 16, DIMC, head * HD, lane);
        cout[j] = wmma32(fa, fb, cout[j]);
      }
    }
    __syncthreads();
  }

  // ---- epilogue: +b_out, +residual x, back-shift, write x1 to d_out ----
  {
    int mt  = wv >> 1;
    int ntb = (wv & 1) * 6;
#pragma unroll
    for (int j = 0; j < 6; ++j) {
      int n = (ntb + j) * 16 + (lane & 15);
      float bn = b_out[n];
#pragma unroll
      for (int r = 0; r < 8; ++r) {
        int m = mt * 16 + r + half * 8;
        if (m < WS2) {
          int rr = m / WSZ, cc = m % WSZ;
          int h = (wy * WSZ + rr + 3) % HW;
          int w = (wx * WSZ + cc + 3) % HW;
          size_t g = ((size_t)((b * HW + h) * HW + w)) * DIMC + n;
          out[g] = x[g] + cout[j][r] + bn;
        }
      }
    }
  }
}

// ---------------- fused MLP: LN2 + FC1 + GELU(exact) + FC2 + residual --------
__global__ __launch_bounds__(256) void swin_mlp_kernel(
    const _Float16* __restrict__ w1T, const _Float16* __restrict__ w2T,
    const float* __restrict__ b1, const float* __restrict__ b2,
    const float* __restrict__ ln_g, const float* __restrict__ ln_b,
    float* __restrict__ out) {
  __shared__ __align__(16) _Float16 sA[32 * 200];  // LN2(x1), f16
  __shared__ __align__(16) _Float16 sH[32 * 776];  // hidden (768) after GELU, f16
  const int tid  = threadIdx.x;
  const int lane = tid & 31;
  const int wv   = tid >> 5;
  const int half = lane >> 4;
  const size_t tok0 = (size_t)blockIdx.x * 32;

  if (tid < 32) {
    const float* px = out + (tok0 + tid) * DIMC;
    float mu = 0.f;
#pragma unroll
    for (int k = 0; k < DIMC; k += 4) {
      float4 v = *(const float4*)(px + k);
      mu += v.x + v.y + v.z + v.w;
    }
    mu *= (1.f / DIMC);
    float var = 0.f;
#pragma unroll
    for (int k = 0; k < DIMC; k += 4) {
      float4 v = *(const float4*)(px + k);
      float a0 = v.x - mu, a1 = v.y - mu, a2 = v.z - mu, a3 = v.w - mu;
      var += a0 * a0 + a1 * a1 + a2 * a2 + a3 * a3;
    }
    var *= (1.f / DIMC);
    float rs = rsqrtf(var + EPSV);
    for (int k = 0; k < DIMC; ++k)
      sA[tid * 200 + k] = (_Float16)((px[k] - mu) * rs * ln_g[k] + ln_b[k]);
  }
  __syncthreads();

  // FC1 + GELU: 2 Mtiles x 48 Ntiles = 96 wave-tiles, 12 per wave
#pragma unroll 1
  for (int i = 0; i < 12; ++i) {
    int task = wv + i * 8;
    int mt = task & 1, nt = task >> 1;
    v8f acc = vzero8();
#pragma unroll
    for (int ks = 0; ks < 6; ++ks) {
      v16h fa = frag_rm(sA, mt * 16, 200, ks * 32, lane);
      v16h fb = frag_rm(w1T, nt * 16, DIMC, ks * 32, lane);
      acc = wmma32(fa, fb, acc);
    }
    int n = nt * 16 + (lane & 15);
    float bn = b1[n];
#pragma unroll
    for (int r = 0; r < 8; ++r) {
      int m = mt * 16 + r + half * 8;
      float v = acc[r] + bn;
      float g = 0.5f * v * (1.f + erff(v * 0.70710678118654752f));
      sH[m * 776 + n] = (_Float16)g;
    }
  }
  __syncthreads();

  // FC2 + residual: 2 Mtiles x 12 Ntiles = 24 wave-tiles, 3 per wave
#pragma unroll 1
  for (int i = 0; i < 3; ++i) {
    int task = wv + i * 8;
    int mt = task & 1, nt = task >> 1;
    v8f acc = vzero8();
#pragma unroll
    for (int ks = 0; ks < 24; ++ks) {
      v16h fa = frag_rm(sH, mt * 16, 776, ks * 32, lane);
      v16h fb = frag_rm(w2T, nt * 16, MLPD, ks * 32, lane);
      acc = wmma32(fa, fb, acc);
    }
    int n = nt * 16 + (lane & 15);
    float bn = b2[n];
#pragma unroll
    for (int r = 0; r < 8; ++r) {
      int m = mt * 16 + r + half * 8;
      size_t g = (tok0 + m) * DIMC + n;
      out[g] = out[g] + acc[r] + bn;
    }
  }
}

extern "C" void kernel_launch(void* const* d_in, const int* in_sizes, int n_in,
                              void* d_out, int out_size, void* d_ws, size_t ws_size,
                              hipStream_t stream) {
  const float* x      = (const float*)d_in[0];
  const float* w_qkv  = (const float*)d_in[1];
  const float* w_out  = (const float*)d_in[2];
  const float* b_out  = (const float*)d_in[3];
  const float* pos    = (const float*)d_in[4];
  const float* ln1_g  = (const float*)d_in[5];
  const float* ln1_b  = (const float*)d_in[6];
  const float* ln2_g  = (const float*)d_in[7];
  const float* ln2_b  = (const float*)d_in[8];
  const float* w1     = (const float*)d_in[9];
  const float* b1     = (const float*)d_in[10];
  const float* w2     = (const float*)d_in[11];
  const float* b2     = (const float*)d_in[12];
  float* out = (float*)d_out;

  char* ws = (char*)d_ws;
  _Float16* wqkvT  = (_Float16*)(ws);                 // 576x192 f16 = 221184 B
  _Float16* woutT  = (_Float16*)(ws + 221184);        // 192x192 f16 =  73728 B
  _Float16* w1T    = (_Float16*)(ws + 294912);        // 768x192 f16 = 294912 B
  _Float16* w2T    = (_Float16*)(ws + 589824);        // 192x768 f16 = 294912 B
  float*    bias49 = (float*)(ws + 884736);           // 49x49 f32   =   9604 B

  int preptot = 576 * 192 + 192 * 192 + 768 * 192 + 192 * 768 + WS2 * WS2;
  swin_prep_kernel<<<(preptot + 255) / 256, 256, 0, stream>>>(
      w_qkv, w_out, w1, w2, pos, wqkvT, woutT, w1T, w2T, bias49);

  // 8 batches x 16x16 windows = 2048 blocks
  swin_attn_kernel<<<2048, 256, 0, stream>>>(x, wqkvT, woutT, b_out, bias49,
                                             ln1_g, ln1_b, out);

  // 100352 tokens / 32 per block = 3136 blocks
  swin_mlp_kernel<<<3136, 256, 0, stream>>>(w1T, w2T, b1, b2, ln2_g, ln2_b, out);
}
